// TexturizeLabels_84267258347944
// MI455X (gfx1250) — matrix operand order repaired and checked by reading, not compiled, not run on hardware
//
#include <hip/hip_runtime.h>

// ---------------------------------------------------------------------------
// TexturizeLabels: 64-bin segmented mean + per-label shift.
//   Pass 0: zero 128-float scratch (sums[64], counts[64]).
//   Pass 1: one-hot WMMA segment reduction (v_wmma_f32_16x16x32_f16).
//           B128 vectorized: each wave streams 128 elements/trip, 4 WMMA
//           sub-chunks, no per-iteration branching (bounds hoisted to host).
//   Pass 2: delta[64] = sum/max(cnt,1) - intensity; out = tex - (lab>0)*delta.
// Memory-bound: ~503 MB total traffic -> ~21.6 us floor at 23.3 TB/s.
// ---------------------------------------------------------------------------

typedef __attribute__((ext_vector_type(16))) _Float16 v16h;
typedef __attribute__((ext_vector_type(8)))  float    v8f;

union H16 { _Float16 h; unsigned short u; };
__device__ inline unsigned short h_bits(_Float16 h) { H16 x; x.h = h; return x.u; }
__device__ inline _Float16 bits_h(unsigned short u) { H16 x; x.u = u; return x.h; }

__global__ void zero_ws_kernel(float* ws) {
    ws[threadIdx.x] = 0.0f;   // 128 threads: sums[64] + counts[64]
}

// Pack one element for the LDS stage: {hi|lo f16 pair, label}.
__device__ inline unsigned long long pack_elem(float v, unsigned l) {
    _Float16 hi = (_Float16)v;
    _Float16 lo = (_Float16)(v - (float)hi);
    unsigned hilo = (unsigned)h_bits(hi) | ((unsigned)h_bits(lo) << 16);
    return ((unsigned long long)hilo << 32) | (unsigned long long)l;
}

// One 16x16x32 one-hot accumulation step over a 32-element stage group.
//   A[m,k] = (lab_k>>4 == m) ? val_k : 0 ; B[k,n] = (lab_k&15 == n)
//   D[m,n] += sum over label 16m+n.  (m = lane&15, h = lane>>4)
__device__ inline void wmma_accum(const unsigned long long* st, unsigned m, unsigned h,
                                  v8f& acc_hi, v8f& acc_lo, v8f& acc_ct) {
    v16h a_hi, a_lo, a_ct, bmat;
#pragma unroll
    for (int j = 0; j < 16; ++j) {
        // A fragment K index for half j (contiguous k order 16h + j, same as B)
        unsigned kA = 16u * h + (unsigned)j;
        // B fragment K index (contiguous k order, same as A)
        unsigned kB = 16u * h + (unsigned)j;

        unsigned long long ea = st[kA];   // uniform per half-wave: LDS broadcast
        unsigned long long eb = st[kB];
        unsigned labA = (unsigned)ea & 63u;
        unsigned hl   = (unsigned)(ea >> 32);
        unsigned labB = (unsigned)eb & 63u;

        bool rm = ((labA >> 4) == m);
        a_hi[j] = rm ? bits_h((unsigned short)(hl & 0xFFFFu)) : (_Float16)0;
        a_lo[j] = rm ? bits_h((unsigned short)(hl >> 16))     : (_Float16)0;
        a_ct[j] = rm ? (_Float16)1 : (_Float16)0;
        bmat[j] = ((labB & 15u) == m) ? (_Float16)1 : (_Float16)0;
    }
    acc_hi = __builtin_amdgcn_wmma_f32_16x16x32_f16(false, a_hi, false, bmat,
                                                    (short)0, acc_hi, false, false);
    acc_lo = __builtin_amdgcn_wmma_f32_16x16x32_f16(false, a_lo, false, bmat,
                                                    (short)0, acc_lo, false, false);
    acc_ct = __builtin_amdgcn_wmma_f32_16x16x32_f16(false, a_ct, false, bmat,
                                                    (short)0, acc_ct, false, false);
}

__global__ __launch_bounds__(256) void seg_reduce_wmma_kernel(
    const float* __restrict__ tex, const int* __restrict__ lab,
    float* __restrict__ g_sum, float* __restrict__ g_cnt, int n, int nfull)
{
    __shared__ unsigned long long stage[8][128];  // per-wave {hilo, label} x128
    __shared__ float bsum[64];
    __shared__ float bcnt[64];

    const int tid  = threadIdx.x;
    const int lane = tid & 31;
    const int wv   = tid >> 5;
    if (tid < 64) { bsum[tid] = 0.0f; bcnt[tid] = 0.0f; }
    __syncthreads();

    const unsigned m = (unsigned)lane & 15u;       // A-row / B-col for this lane
    const unsigned h = ((unsigned)lane >> 4) & 1u; // half-wave selector

    v8f acc_hi = {}; v8f acc_lo = {}; v8f acc_ct = {};

    const int wavesPerBlk = blockDim.x >> 5;
    const int totalWaves  = gridDim.x * wavesPerBlk;
    const int gwave       = blockIdx.x * wavesPerBlk + wv;

    // ---- main loop: 128 contiguous elements/wave/trip, B128 loads, no branches
    const int stride = totalWaves * 128;
    for (int base = gwave * 128; base < nfull; base += stride) {
        const float4 v4 = ((const float4*)tex)[(base >> 2) + lane];
        const int4   l4 = ((const int4*)lab)[(base >> 2) + lane];
        __builtin_prefetch((const char*)(tex + base + lane * 4) + (size_t)stride * 4, 0, 1);
        __builtin_prefetch((const char*)(lab + base + lane * 4) + (size_t)stride * 4, 0, 1);

        stage[wv][4 * lane + 0] = pack_elem(v4.x, (unsigned)l4.x & 63u);
        stage[wv][4 * lane + 1] = pack_elem(v4.y, (unsigned)l4.y & 63u);
        stage[wv][4 * lane + 2] = pack_elem(v4.z, (unsigned)l4.z & 63u);
        stage[wv][4 * lane + 3] = pack_elem(v4.w, (unsigned)l4.w & 63u);
        __asm__ volatile("s_wait_dscnt 0" ::: "memory");  // same-wave LDS RAW

#pragma unroll
        for (int s = 0; s < 4; ++s)
            wmma_accum(&stage[wv][32 * s], m, h, acc_hi, acc_lo, acc_ct);
    }

    // ---- tail: predicated 32-element chunks (empty for the harness shape)
    for (int base = nfull + gwave * 32; base < n; base += totalWaves * 32) {
        int idx = base + lane;
        float v = 0.0f; unsigned l = 0u;
        if (idx < n) { v = tex[idx]; l = (unsigned)lab[idx] & 63u; }
        stage[wv][lane] = pack_elem(v, l);
        __asm__ volatile("s_wait_dscnt 0" ::: "memory");
        wmma_accum(&stage[wv][0], m, h, acc_hi, acc_lo, acc_ct);
    }

    // D layout: lanes 0-15, VGPR r -> (M=r, N=lane). Labels 0..63 live in M=0..3.
    if (lane < 16) {
#pragma unroll
        for (int r = 0; r < 4; ++r) {
            int bin = r * 16 + lane;
            atomicAdd(&bsum[bin], acc_hi[r] + acc_lo[r]);  // ds_add_f32
            atomicAdd(&bcnt[bin], acc_ct[r]);
        }
    }
    __syncthreads();
    if (tid < 64) {
        atomicAdd(&g_sum[tid], bsum[tid]);  // global_atomic_add_f32
        atomicAdd(&g_cnt[tid], bcnt[tid]);
    }
}

// ---------------------------------------------------------------------------
// Pass 2: streaming apply, 128-bit vectorized.
// ---------------------------------------------------------------------------
__global__ __launch_bounds__(256) void apply_kernel(
    const float4* __restrict__ tex, const int4* __restrict__ lab,
    const float* __restrict__ g_sum, const float* __restrict__ g_cnt,
    const float* __restrict__ inten, float4* __restrict__ out, int n4)
{
    __shared__ float sdelta[64];
    if (threadIdx.x < 64) {
        float c = g_cnt[threadIdx.x];
        float mean = g_sum[threadIdx.x] / fmaxf(c, 1.0f);
        sdelta[threadIdx.x] = mean - inten[threadIdx.x];
    }
    __syncthreads();

    const int stride = gridDim.x * blockDim.x;
    for (int i = blockIdx.x * blockDim.x + threadIdx.x; i < n4; i += stride) {
        __builtin_prefetch((const char*)(tex + i) + (size_t)stride * 16, 0, 1);
        __builtin_prefetch((const char*)(lab + i) + (size_t)stride * 16, 0, 1);
        float4 v = tex[i];
        int4   l = lab[i];
        float4 o;
        o.x = v.x - (l.x > 0 ? sdelta[l.x & 63] : 0.0f);
        o.y = v.y - (l.y > 0 ? sdelta[l.y & 63] : 0.0f);
        o.z = v.z - (l.z > 0 ? sdelta[l.z & 63] : 0.0f);
        o.w = v.w - (l.w > 0 ? sdelta[l.w & 63] : 0.0f);
        out[i] = o;
    }
}

extern "C" void kernel_launch(void* const* d_in, const int* in_sizes, int n_in,
                              void* d_out, int out_size, void* d_ws, size_t ws_size,
                              hipStream_t stream) {
    const float* tex   = (const float*)d_in[0];
    const int*   lab   = (const int*)d_in[1];
    const float* inten = (const float*)d_in[2];
    float*       out   = (float*)d_out;
    const int    n     = in_sizes[0];

    float* g_sum = (float*)d_ws;       // [64]
    float* g_cnt = g_sum + 64;         // [64]

    hipLaunchKernelGGL(zero_ws_kernel, dim3(1), dim3(128), 0, stream, g_sum);

    // 1024 blocks x 256 threads = 8192 waves; trip = 8192*128 = 1,048,576 elems.
    // n = 25,165,824 = 24 full trips exactly -> nfull == n, tail loop empty.
    const int blocksR    = 1024;
    const int totalWaves = blocksR * (256 / 32);
    const int trip       = totalWaves * 128;
    const int nfull      = n - (n % trip);
    hipLaunchKernelGGL(seg_reduce_wmma_kernel, dim3(blocksR), dim3(256), 0, stream,
                       tex, lab, g_sum, g_cnt, n, nfull);

    const int n4 = n >> 2;
    hipLaunchKernelGGL(apply_kernel, dim3(2048), dim3(256), 0, stream,
                       (const float4*)tex, (const int4*)lab,
                       g_sum, g_cnt, inten, (float4*)out, n4);
}
